// GraphCore_33054068310578
// MI455X (gfx1250) — matrix operand order repaired and checked by reading, not compiled, not run on hardware
//
#include <hip/hip_runtime.h>

// ---------------------------------------------------------------------------
// GraphCore (edge MLP + scatter-mean + node MLP + residuals) for gfx1250.
// GEMMs via v_wmma_f32_16x16x32_f16 (f16 inputs, f32 accumulation).
// Weights are staged through a double-buffered LDS slab, cooperatively
// prefetched by the whole block so WMMAs never wait on global loads.
// ---------------------------------------------------------------------------

typedef __attribute__((ext_vector_type(16))) _Float16 v16h;
typedef __attribute__((ext_vector_type(8)))  _Float16 v8h;
typedef __attribute__((ext_vector_type(4)))  _Float16 v4h;
typedef __attribute__((ext_vector_type(8)))  float    v8f;

#define SLAB_LD 40  // halves per slab row (32 data + 8 pad, 80B keeps 16B align)

// ---- WMMA fragment loaders (contiguous-k order k = 16*hi + s, same for A and B, wave32) -------------------
// A 16x32 f16: lane m=lane&15, hi=lane>=16; K elements hi*16 .. hi*16+15
// -> two 16B loads.
__device__ __forceinline__ v16h load_frag_a(const _Float16* p) {
  v8h lo = *(const v8h*)(p);
  v8h hi = *(const v8h*)(p + 8);
  return __builtin_shufflevector(lo, hi, 0, 1, 2, 3, 4, 5, 6, 7,
                                         8, 9, 10, 11, 12, 13, 14, 15);
}
// B 32x16 f16: lane n=lane&15; K elements hi*16 .. hi*16+15 -> two 16B loads.
__device__ __forceinline__ v16h load_frag_b(const _Float16* p) {
  v8h lo = *(const v8h*)(p);
  v8h hi = *(const v8h*)(p + 8);
  return __builtin_shufflevector(lo, hi, 0, 1, 2, 3, 4, 5, 6, 7,
                                         8, 9, 10, 11, 12, 13, 14, 15);
}

__device__ __forceinline__ v8f wmma_f16(v16h a, v16h b, v8f c) {
  return __builtin_amdgcn_wmma_f32_16x16x32_f16(
      /*neg_a=*/false, a, /*neg_b=*/false, b,
      /*c_mod=*/(short)0, c, /*reuse_a=*/false, /*reuse_b=*/false);
}

// ---------------------------------------------------------------------------
// Block-cooperative GEMM: each of the 4 waves owns a 16-row M-tile of A
// (LDS, row stride K) and accumulates a 16 x (NT*16) tile.  The weight slab
// for each 32-wide K step lives in LDS ([NOUT][SLAB_LD] f16, double buffered);
// the next slab is prefetched into registers before the WMMA block and
// drained to LDS after it, so global latency overlaps the matrix pipe.
// Wt: global f16, layout [Nout][K] (pre-transposed weights).
// ---------------------------------------------------------------------------
template<int K, int NT>
__device__ __forceinline__ void block_gemm(const _Float16* aWave,
                                           const _Float16* __restrict__ Wt,
                                           _Float16* slab0, _Float16* slab1,
                                           int tid, int lane, v8f (&acc)[NT]) {
  static_assert(NT >= 2 && (NT % 2) == 0, "NT must be even");
  constexpr int KT   = K / 32;
  constexpr int NOUT = NT * 16;
  constexpr int CH   = NOUT / 32;  // 16B chunks per thread (128 thr, 4/row)
  const int hi = lane >> 4;        // 0 or 1
  const int ln = lane & 15;        // row (A) / column (B)
  const _Float16* aRow = aWave + ln * K + hi * 16;

  // Prologue: block loads slab for kt=0.
#pragma unroll
  for (int i = 0; i < CH; ++i) {
    const int c = tid + i * 128;
    const int n = c >> 2, part = c & 3;
    *(v8h*)(slab0 + n * SLAB_LD + part * 8) =
        *(const v8h*)(Wt + (size_t)n * K + part * 8);
  }
  __syncthreads();

  for (int kt = 0; kt < KT; ++kt) {
    _Float16* cur = (kt & 1) ? slab1 : slab0;
    _Float16* nxt = (kt & 1) ? slab0 : slab1;
    const int k0 = kt * 32;
    const bool haveNext = (kt + 1 < KT);

    // Issue global prefetch of next slab (results land after the WMMA block).
    v8h pf[CH];
    if (haveNext) {
#pragma unroll
      for (int i = 0; i < CH; ++i) {
        const int c = tid + i * 128;
        const int n = c >> 2, part = c & 3;
        pf[i] = *(const v8h*)(Wt + (size_t)n * K + (k0 + 32) + part * 8);
      }
    }

    // NT WMMAs fed from LDS (depth-2 pipelined B stream).
    v16h a  = load_frag_a(aRow + k0);
    v16h b0 = load_frag_b(cur + ln * SLAB_LD + hi * 16);
    v16h b1 = load_frag_b(cur + (16 + ln) * SLAB_LD + hi * 16);
#pragma unroll
    for (int nt = 0; nt < NT; nt += 2) {
      v16h bn0 = b0, bn1 = b1;
      if (nt + 2 < NT) {
        bn0 = load_frag_b(cur + ((nt + 2) * 16 + ln) * SLAB_LD + hi * 16);
        bn1 = load_frag_b(cur + ((nt + 3) * 16 + ln) * SLAB_LD + hi * 16);
      }
      acc[nt]     = wmma_f16(a, b0, acc[nt]);
      acc[nt + 1] = wmma_f16(a, b1, acc[nt + 1]);
      b0 = bn0;
      b1 = bn1;
    }

    // Drain prefetched slab to the other buffer; barrier flips buffers.
    if (haveNext) {
#pragma unroll
      for (int i = 0; i < CH; ++i) {
        const int c = tid + i * 128;
        const int n = c >> 2, part = c & 3;
        *(v8h*)(nxt + n * SLAB_LD + part * 8) = pf[i];
      }
    }
    __syncthreads();
  }
}

// Full hidden layer: GEMM + bias + ReLU, result stored to LDS as f16.
// oWave: LDS, this wave's 16 rows, row stride = NOUT halves.
template<int K, int NOUT>
__device__ __forceinline__ void block_layer(const _Float16* aWave,
                                            const _Float16* __restrict__ Wt,
                                            const float* __restrict__ bias,
                                            _Float16* oWave,
                                            _Float16* slab0, _Float16* slab1,
                                            int tid, int lane) {
  constexpr int NT = NOUT / 16;
  v8f acc[NT] = {};
  block_gemm<K, NT>(aWave, Wt, slab0, slab1, tid, lane, acc);
  const int hi = lane >> 4;
  const int n  = lane & 15;
#pragma unroll
  for (int nt = 0; nt < NT; ++nt) {
#pragma unroll
    for (int r = 0; r < 8; ++r) {
      const int m  = r + hi * 8;          // C/D layout: VGPR r -> row r + hi*8
      const int nn = nt * 16 + n;
      float v = acc[nt][r] + bias[nn];
      v = fmaxf(v, 0.0f);
      oWave[m * NOUT + nn] = (_Float16)v;
    }
  }
}

// ---------------------------------------------------------------------------
// Edge MLP: in = cat[x[row], x[col], edge_attr] (384) -> 256 -> 256 -> 256 -> 128
// Fused epilogue: out_edge = edge_attr + e_out ; atomic scatter-add of e_out.
// Block: 128 threads (4 waves), 64 edges.
// LDS halves: ein 24576 | act1 16384 | slab0 10240 | slab1 10240 = 122880 B.
// ---------------------------------------------------------------------------
__global__ __launch_bounds__(128) void edge_mlp_kernel(
    const float* __restrict__ x, const int* __restrict__ eidx,
    const float* __restrict__ eattr,
    const _Float16* __restrict__ Wt1, const float* __restrict__ b1,
    const _Float16* __restrict__ Wt2, const float* __restrict__ b2,
    const _Float16* __restrict__ Wt3, const float* __restrict__ b3,
    const _Float16* __restrict__ Wt4, const float* __restrict__ b4,
    float* __restrict__ out_edge, float* __restrict__ s_accum,
    int E, int N) {
  extern __shared__ _Float16 smem[] __attribute__((aligned(64)));
  __shared__ int rowv[64];
  __shared__ int colv[64];

  const int tid  = threadIdx.x;
  const int lane = tid & 31;
  const int wave = tid >> 5;
  const int e0   = blockIdx.x * 64;

  _Float16* ein   = smem;                      // [64][384]
  _Float16* act1  = smem + 24576;              // [64][256]
  _Float16* slab0 = smem + 24576 + 16384;      // [256][SLAB_LD]
  _Float16* slab1 = slab0 + 256 * SLAB_LD;

  if (tid < 64) {
    int e = e0 + tid;
    e = (e < E) ? e : (E - 1);
    rowv[tid] = eidx[e];
    colv[tid] = eidx[E + e];
  }
  __syncthreads();

  // Cooperative gather: edge_in[64][384] f16 (chunks of 4 floats).
  for (int t = tid; t < 64 * 96; t += 128) {
    const int m = t / 96;
    const int c = t - m * 96;
    int e = e0 + m;
    e = (e < E) ? e : (E - 1);
    const float* src;
    if (c < 32)       src = x + (size_t)rowv[m] * 128 + c * 4;
    else if (c < 64)  src = x + (size_t)colv[m] * 128 + (c - 32) * 4;
    else              src = eattr + (size_t)e * 128 + (c - 64) * 4;
    const float4 v = *(const float4*)src;
    const v4h h = { (_Float16)v.x, (_Float16)v.y, (_Float16)v.z, (_Float16)v.w };
    *(v4h*)(&ein[m * 384 + c * 4]) = h;
  }
  __syncthreads();

  // Per-wave activation tiles (16 rows each).
  _Float16* einW  = ein + wave * 16 * 384;     // layer-1 A, stride 384
  _Float16* act1W = act1 + wave * 16 * 256;    // stride 256
  _Float16* act2W = ein + wave * 16 * 384;     // overlay on dead ein rows

  block_layer<384, 256>(einW,  Wt1, b1, act1W, slab0, slab1, tid, lane);
  block_layer<256, 256>(act1W, Wt2, b2, act2W, slab0, slab1, tid, lane);
  block_layer<256, 256>(act2W, Wt3, b3, act1W, slab0, slab1, tid, lane);

  // Layer 4: K=256, N=128, fused residual + scatter-add epilogue.
  v8f acc[8] = {};
  block_gemm<256, 8>(act1W, Wt4, slab0, slab1, tid, lane, acc);
  const int hi = lane >> 4;
  const int n  = lane & 15;
#pragma unroll
  for (int nt = 0; nt < 8; ++nt) {
#pragma unroll
    for (int r = 0; r < 8; ++r) {
      const int m = wave * 16 + r + hi * 8;
      const int e = e0 + m;
      if (e < E) {
        const int f = nt * 16 + n;
        const float v = acc[nt][r] + b4[f];
        const size_t oi = (size_t)e * 128 + f;
        out_edge[oi] = eattr[oi] + v;
        atomicAdd(&s_accum[(size_t)colv[m] * 128 + f], v);
      }
    }
  }
}

// ---------------------------------------------------------------------------
// Node MLP: in = cat[x, agg] (256) -> 256 -> 256 -> 256 -> 128 ; out = x + n_out
// agg = scatter_sum / max(count, 1).
// LDS halves: nin 16384 | act1 16384 | slab0 10240 | slab1 10240 = 106496 B.
// ---------------------------------------------------------------------------
__global__ __launch_bounds__(128) void node_mlp_kernel(
    const float* __restrict__ x, const float* __restrict__ s_accum,
    const float* __restrict__ cnt,
    const _Float16* __restrict__ Wt1, const float* __restrict__ b1,
    const _Float16* __restrict__ Wt2, const float* __restrict__ b2,
    const _Float16* __restrict__ Wt3, const float* __restrict__ b3,
    const _Float16* __restrict__ Wt4, const float* __restrict__ b4,
    float* __restrict__ out_node, int N) {
  extern __shared__ _Float16 smem[] __attribute__((aligned(64)));
  __shared__ float invc[64];

  const int tid  = threadIdx.x;
  const int lane = tid & 31;
  const int wave = tid >> 5;
  const int n0   = blockIdx.x * 64;

  _Float16* nin   = smem;                      // [64][256]
  _Float16* act1  = smem + 16384;              // [64][256]
  _Float16* slab0 = smem + 32768;              // [256][SLAB_LD]
  _Float16* slab1 = slab0 + 256 * SLAB_LD;

  if (tid < 64) {
    int node = n0 + tid;
    node = (node < N) ? node : (N - 1);
    invc[tid] = 1.0f / fmaxf(cnt[node], 1.0f);
  }
  __syncthreads();

  for (int t = tid; t < 64 * 64; t += 128) {
    const int m = t >> 6;
    const int c = t & 63;
    int node = n0 + m;
    node = (node < N) ? node : (N - 1);
    float4 v;
    if (c < 32) {
      v = *(const float4*)(x + (size_t)node * 128 + c * 4);
    } else {
      const float4 sv = *(const float4*)(s_accum + (size_t)node * 128 + (c - 32) * 4);
      const float inv = invc[m];
      v.x = sv.x * inv; v.y = sv.y * inv; v.z = sv.z * inv; v.w = sv.w * inv;
    }
    const v4h h = { (_Float16)v.x, (_Float16)v.y, (_Float16)v.z, (_Float16)v.w };
    *(v4h*)(&nin[m * 256 + c * 4]) = h;
  }
  __syncthreads();

  _Float16* ninW  = nin + wave * 16 * 256;     // layer-1 A
  _Float16* act1W = act1 + wave * 16 * 256;
  _Float16* act2W = nin + wave * 16 * 256;     // overlay on dead nin rows

  block_layer<256, 256>(ninW,  Wt1, b1, act1W, slab0, slab1, tid, lane);
  block_layer<256, 256>(act1W, Wt2, b2, act2W, slab0, slab1, tid, lane);
  block_layer<256, 256>(act2W, Wt3, b3, act1W, slab0, slab1, tid, lane);

  v8f acc[8] = {};
  block_gemm<256, 8>(act1W, Wt4, slab0, slab1, tid, lane, acc);
  const int hi = lane >> 4;
  const int n  = lane & 15;
#pragma unroll
  for (int nt = 0; nt < 8; ++nt) {
#pragma unroll
    for (int r = 0; r < 8; ++r) {
      const int m    = wave * 16 + r + hi * 8;
      const int node = n0 + m;
      if (node < N) {
        const int f = nt * 16 + n;
        const size_t oi = (size_t)node * 128 + f;
        out_node[oi] = x[oi] + acc[nt][r] + b4[f];
      }
    }
  }
}

// ---------------------------------------------------------------------------
// Prep kernels: zero accumulators, f32->f16 transposed weights, edge counts.
// ---------------------------------------------------------------------------
__global__ void zero_ws_kernel(float* __restrict__ s, float* __restrict__ cnt,
                               int n_s, int n_c) {
  const int i = blockIdx.x * blockDim.x + threadIdx.x;
  if (i < n_s) s[i] = 0.0f;
  else if (i < n_s + n_c) cnt[i - n_s] = 0.0f;
}

// W: f32 row-major [K][Nout]  ->  Wt: f16 row-major [Nout][K]
__global__ void convert_wt_kernel(const float* __restrict__ W,
                                  _Float16* __restrict__ Wt, int K, int Nout) {
  const int i = blockIdx.x * blockDim.x + threadIdx.x;
  if (i < K * Nout) {
    const int n = i / K;
    const int k = i - n * K;
    Wt[i] = (_Float16)W[k * Nout + n];
  }
}

__global__ void count_edges_kernel(const int* __restrict__ eidx,
                                   float* __restrict__ cnt, int E) {
  const int e = blockIdx.x * blockDim.x + threadIdx.x;
  if (e < E) atomicAdd(&cnt[eidx[E + e]], 1.0f);
}

// ---------------------------------------------------------------------------
extern "C" void kernel_launch(void* const* d_in, const int* in_sizes, int n_in,
                              void* d_out, int out_size, void* d_ws, size_t ws_size,
                              hipStream_t stream) {
  (void)n_in; (void)out_size; (void)ws_size;
  const float* x     = (const float*)d_in[0];
  const int*   eidx  = (const int*)d_in[1];
  const float* eattr = (const float*)d_in[2];
  // d_in[3] = u, d_in[4] = batch (unused by the module)
  const float* eW[4] = {(const float*)d_in[5],  (const float*)d_in[7],
                        (const float*)d_in[9],  (const float*)d_in[11]};
  const float* eb[4] = {(const float*)d_in[6],  (const float*)d_in[8],
                        (const float*)d_in[10], (const float*)d_in[12]};
  const float* nW[4] = {(const float*)d_in[13], (const float*)d_in[15],
                        (const float*)d_in[17], (const float*)d_in[19]};
  const float* nb[4] = {(const float*)d_in[14], (const float*)d_in[16],
                        (const float*)d_in[18], (const float*)d_in[20]};

  const int N = in_sizes[0] / 128;  // nodes
  const int E = in_sizes[2] / 128;  // edges

  // Workspace carve-out (256B aligned).
  char* ws = (char*)d_ws;
  size_t off = 0;
  auto carve = [&](size_t bytes) -> void* {
    void* p = ws + off;
    off = (off + bytes + 255) & ~(size_t)255;
    return p;
  };
  float* s_accum = (float*)carve((size_t)N * 128 * sizeof(float));
  float* cnt     = (float*)carve((size_t)N * sizeof(float));
  const int eK[4] = {384, 256, 256, 256}, eN[4] = {256, 256, 256, 128};
  const int nK[4] = {256, 256, 256, 256}, nN[4] = {256, 256, 256, 128};
  _Float16* eWt[4];
  _Float16* nWt[4];
  for (int i = 0; i < 4; ++i) eWt[i] = (_Float16*)carve((size_t)eK[i] * eN[i] * 2);
  for (int i = 0; i < 4; ++i) nWt[i] = (_Float16*)carve((size_t)nK[i] * nN[i] * 2);

  // 1) zero scatter accumulators (must be re-done every call; graph replays).
  {
    const int total = N * 128 + N;
    zero_ws_kernel<<<(total + 255) / 256, 256, 0, stream>>>(s_accum, cnt,
                                                            N * 128, N);
  }
  // 2) weight conversion f32 -> transposed f16.
  for (int i = 0; i < 4; ++i) {
    int tot = eK[i] * eN[i];
    convert_wt_kernel<<<(tot + 255) / 256, 256, 0, stream>>>(eW[i], eWt[i],
                                                             eK[i], eN[i]);
    tot = nK[i] * nN[i];
    convert_wt_kernel<<<(tot + 255) / 256, 256, 0, stream>>>(nW[i], nWt[i],
                                                             nK[i], nN[i]);
  }
  // 3) per-node incoming-edge counts.
  count_edges_kernel<<<(E + 255) / 256, 256, 0, stream>>>(eidx, cnt, E);

  float* out_node = (float*)d_out;
  float* out_edge = out_node + (size_t)N * 128;

  // 4) edge MLP + residual + scatter-add.  LDS = 122880 B.
  edge_mlp_kernel<<<(E + 63) / 64, 128, 122880, stream>>>(
      x, eidx, eattr,
      eWt[0], eb[0], eWt[1], eb[1], eWt[2], eb[2], eWt[3], eb[3],
      out_edge, s_accum, E, N);

  // 5) node MLP + residual.  LDS = 106496 B.
  node_mlp_kernel<<<(N + 63) / 64, 128, 106496, stream>>>(
      x, s_accum, cnt,
      nWt[0], nb[0], nWt[1], nb[1], nWt[2], nb[2], nWt[3], nb[3],
      out_node, N);
}
